// MultiHeadAttention_64020782514542
// MI455X (gfx1250) — matrix operand order repaired and checked by reading, not compiled, not run on hardware
//
#include <hip/hip_runtime.h>
#include <stdint.h>

#define BDIM 4
#define TDIM 2048
#define CDIM 1024
#define HDIM 16
#define DDIM 64
#define MROWS (BDIM * TDIM) /* 8192 */

typedef __attribute__((ext_vector_type(16))) __bf16 v16bf;
typedef __attribute__((ext_vector_type(8)))  float  v8f;

union Frag { uint4 u[2]; v16bf v; };

__device__ __forceinline__ unsigned short f2bf(float x) {
  unsigned int u = __float_as_uint(x);
  unsigned int r = u + 0x7FFFu + ((u >> 16) & 1u);  // round-to-nearest-even
  return (unsigned short)(r >> 16);
}

__device__ __forceinline__ float bf2f(unsigned short h) {
  return __uint_as_float((unsigned int)h << 16);
}

__device__ __forceinline__ unsigned int pack2bf(float lo, float hi) {
  return (unsigned int)f2bf(lo) | ((unsigned int)f2bf(hi) << 16);
}

__device__ __forceinline__ v8f wmma_bf16(v16bf a, v16bf b, v8f c) {
  return __builtin_amdgcn_wmma_f32_16x16x32_bf16(false, a, false, b, (short)0, c,
                                                 false, false);
}

// A-fragment (16x32 bf16): lane holds row (rowptr); K chunks at
// (lane>>4)*8 and (lane>>4)*8+16 per ISA layout.
__device__ __forceinline__ v16bf load_a(const unsigned short* rowptr, int k0, int lane) {
  Frag f;
  int o = k0 + ((lane >> 4) << 3);
  f.u[0] = *(const uint4*)(rowptr + o);
  f.u[1] = *(const uint4*)(rowptr + o + 16);
  return f.v;
}

// B-fragment (32x16 bf16): lane holds column (lane&15) = row of transposed
// operand; K chunks at k0 + (lane>>4)*8 and k0 + (lane>>4)*8 + 16.
__device__ __forceinline__ v16bf load_b(const unsigned short* rowptr, int k0, int lane) {
  Frag f;
  int o = k0 + ((lane >> 4) << 3);
  f.u[0] = *(const uint4*)(rowptr + o);
  f.u[1] = *(const uint4*)(rowptr + o + 16);
  return f.v;
}

// gfx1250 async global->LDS copy, 16B per lane, tracked by ASYNCcnt.
__device__ __forceinline__ void async_copy16(unsigned lds_byte, const char* gaddr) {
  asm volatile("global_load_async_to_lds_b128 %0, %1, off"
               :: "v"(lds_byte), "v"(gaddr) : "memory");
}

// ---------------- conversion / layout kernels ----------------

__global__ void __launch_bounds__(256) cvt_f32_bf16(const float* __restrict__ x,
                                                    unsigned short* __restrict__ y, int n) {
  int i = blockIdx.x * 256 + threadIdx.x;
  if (i < n) y[i] = f2bf(x[i]);
}

// Wt[n*1024 + k] = bf16(W[k*1024 + n])
__global__ void __launch_bounds__(256) transpose_cvt(const float* __restrict__ W,
                                                     unsigned short* __restrict__ Wt) {
  int i = blockIdx.x * 256 + threadIdx.x;  // over 1024*1024
  int n = i >> 10, k = i & 1023;
  Wt[i] = f2bf(W[k * 1024 + n]);
}

__global__ void __launch_bounds__(256) concat_bias(const float* __restrict__ bq,
                                                   const float* __restrict__ bk,
                                                   const float* __restrict__ bv,
                                                   float* __restrict__ o) {
  int i = blockIdx.x * 256 + threadIdx.x;
  if (i < 3072) o[i] = (i < 1024) ? bq[i] : ((i < 2048) ? bk[i - 1024] : bv[i - 2048]);
}

// bit j of mb[b*(T/32)+w] = (attention_mask[b*T + w*32 + j] != 0)
__global__ void __launch_bounds__(256) mask_bits(const int* __restrict__ am,
                                                 unsigned int* __restrict__ mb) {
  int i = blockIdx.x * 256 + threadIdx.x;  // B*T/32 = 256 words
  if (i < BDIM * TDIM / 32) {
    const int* p = am + i * 32;
    unsigned int w = 0u;
#pragma unroll
    for (int k = 0; k < 32; k++) w |= (p[k] != 0 ? 1u : 0u) << k;
    mb[i] = w;
  }
}

// qkv [M,3072] bf16 -> Q (pre-scaled by 1/sqrt(D)), K [B,H,T,D]; V -> [B,H,D,T]
__global__ void __launch_bounds__(256) split_heads(const unsigned short* __restrict__ qkv,
                                                   unsigned short* __restrict__ Q,
                                                   unsigned short* __restrict__ Kh,
                                                   unsigned short* __restrict__ Vt) {
  int idx = blockIdx.x * 256 + threadIdx.x;  // over B*H*T*D = 8388608
  int d = idx & 63;
  int t = (idx >> 6) & (TDIM - 1);
  int h = (idx >> 17) & (HDIM - 1);
  int b = idx >> 21;
  size_t src = (size_t)(b * TDIM + t) * 3072 + h * DDIM + d;
  Q[idx]  = f2bf(bf2f(qkv[src]) * 0.125f);  // fold softmax scale into Q
  Kh[idx] = qkv[src + 1024];
  Vt[((size_t)(b * HDIM + h) * DDIM + d) * TDIM + t] = qkv[src + 2048];
}

// ---------------- WMMA GEMM: out[M,N] = A[M,K] * Bt[N,K]^T + bias ----------------

template <bool OUT_BF16>
__global__ void __launch_bounds__(256) gemm_wmma(const unsigned short* __restrict__ A,
                                                 const unsigned short* __restrict__ Bt,
                                                 const float* __restrict__ bias,
                                                 void* __restrict__ out, int K, int ldo) {
  int lane = threadIdx.x & 31;
  int wid  = threadIdx.x >> 5;
  int wm = wid & 1, wn = wid >> 1;
  int row0 = blockIdx.x * 64 + wm * 32;
  int col0 = blockIdx.y * 256 + wn * 64;

  v8f acc[2][4];
#pragma unroll
  for (int i = 0; i < 2; i++)
#pragma unroll
    for (int j = 0; j < 4; j++) acc[i][j] = (v8f){0, 0, 0, 0, 0, 0, 0, 0};

  const unsigned short* arow[2];
#pragma unroll
  for (int i = 0; i < 2; i++) arow[i] = A + (size_t)(row0 + i * 16 + (lane & 15)) * K;
  const unsigned short* brow[4];
#pragma unroll
  for (int j = 0; j < 4; j++) brow[j] = Bt + (size_t)(col0 + j * 16 + (lane & 15)) * K;

  for (int k0 = 0; k0 < K; k0 += 32) {
    v16bf a[2], b[4];
#pragma unroll
    for (int i = 0; i < 2; i++) {
      a[i] = load_a(arow[i], k0, lane);
      __builtin_prefetch(arow[i] + k0 + 128, 0, 1);  // global_prefetch_b8
    }
#pragma unroll
    for (int j = 0; j < 4; j++) {
      b[j] = load_b(brow[j], k0, lane);
      __builtin_prefetch(brow[j] + k0 + 128, 0, 1);
    }
#pragma unroll
    for (int i = 0; i < 2; i++)
#pragma unroll
      for (int j = 0; j < 4; j++) acc[i][j] = wmma_bf16(a[i], b[j], acc[i][j]);
  }

  int rowoff = (lane >> 4) << 3;  // lanes 16-31 hold rows +8
#pragma unroll
  for (int i = 0; i < 2; i++)
#pragma unroll
    for (int j = 0; j < 4; j++)
#pragma unroll
      for (int r = 0; r < 8; r++) {
        int row = row0 + i * 16 + r + rowoff;
        int col = col0 + j * 16 + (lane & 15);
        float v = acc[i][j][r] + bias[col];
        if (OUT_BF16)
          ((unsigned short*)out)[(size_t)row * ldo + col] = f2bf(v);
        else
          ((float*)out)[(size_t)row * ldo + col] = v;
      }
}

// ---------------- flash attention (transposed-score, async-staged K/V) ----------
// Block = 8 waves = 8 consecutive 16-query tiles of ONE head. K/V 32-key blocks
// are staged into double-buffered LDS with global_load_async_to_lds_b128 (wave 0
// streams K, wave 1 streams V^T), drained with s_wait_asynccnt. All waves run to
// the block's causal bound; out-of-range blocks are neutralized by the validity
// bitmask (sum contribution 0).

__global__ void __launch_bounds__(256) flash_attn(const unsigned short* __restrict__ Qh,
                                                  const unsigned short* __restrict__ Kh,
                                                  const unsigned short* __restrict__ Vt,
                                                  const unsigned int* __restrict__ mbits,
                                                  unsigned short* __restrict__ yatt) {
  __shared__ unsigned short Ktile[2][32 * 64];  // 2 x 4KB, [key][d]
  __shared__ unsigned short Vtile[2][64 * 32];  // 2 x 4KB, [d][key]
  __shared__ unsigned short Plds[8][16 * 32];   // per-wave P tile [q][key]

  int lane = threadIdx.x & 31;
  int wid  = threadIdx.x >> 5;
  int bh = blockIdx.x >> 4;                 // block-uniform head
  int qb0 = (blockIdx.x & 15) << 7;         // first query row of this block
  int q0 = qb0 + (wid << 4);                // this wave's query tile
  int b = bh >> 4, h = bh & 15;
  int lq  = lane & 15;
  int hi8 = (lane >> 4) << 3;

  size_t head = (size_t)bh * TDIM * DDIM;
  const char* Kglob = (const char*)(Kh + head);              // block jb: 4KB @ jb*4096
  const char* Vglob = (const char*)(Vt + (size_t)bh * DDIM * TDIM);

  // Q as B-operand (column q = Q row q0+lq), two 32-wide k-chunks over D
  const unsigned short* Qrow = Qh + head + (size_t)(q0 + lq) * DDIM;
  v16bf qb_0 = load_b(Qrow, 0, lane);
  v16bf qb_1 = load_b(Qrow, 32, lane);

  const unsigned int* mrow = mbits + b * (TDIM / 32);

  float m_cur = -3.0e38f, l_cur = 0.0f;
  v8f acc[4];
#pragma unroll
  for (int c = 0; c < 4; c++) acc[c] = (v8f){0, 0, 0, 0, 0, 0, 0, 0};

  int q_abs = q0 + lq;
  int jmax = (qb0 + 159) >> 5;  // block-uniform causal bound (covers qb0+127)

  unsigned kt0 = (unsigned)(uintptr_t)&Ktile[0][0];
  unsigned kt1 = (unsigned)(uintptr_t)&Ktile[1][0];
  unsigned vt0 = (unsigned)(uintptr_t)&Vtile[0][0];
  unsigned vt1 = (unsigned)(uintptr_t)&Vtile[1][0];

  // prologue: stage block 0
  if (wid == 0) {
#pragma unroll
    for (int it = 0; it < 8; it++)
      async_copy16(kt0 + it * 512 + lane * 16, Kglob + it * 512 + lane * 16);
  } else if (wid == 1) {
#pragma unroll
    for (int it = 0; it < 8; it++) {
      int chunk = it * 32 + lane;
      int d = chunk >> 2, sub = chunk & 3;
      async_copy16(vt0 + chunk * 16, Vglob + ((size_t)d * TDIM) * 2 + sub * 16);
    }
  }

  for (int jb = 0; jb < jmax; jb++) {
    int key0 = jb << 5;
    const unsigned short* Kt = (jb & 1) ? &Ktile[1][0] : &Ktile[0][0];
    const unsigned short* Vx = (jb & 1) ? &Vtile[1][0] : &Vtile[0][0];

    // stage next block into the other buffer
    if (jb + 1 < jmax) {
      int nk = (jb + 1) << 5;
      unsigned ktn = ((jb + 1) & 1) ? kt1 : kt0;
      unsigned vtn = ((jb + 1) & 1) ? vt1 : vt0;
      if (wid == 0) {
#pragma unroll
        for (int it = 0; it < 8; it++)
          async_copy16(ktn + it * 512 + lane * 16,
                       Kglob + (size_t)nk * 128 + it * 512 + lane * 16);
      } else if (wid == 1) {
#pragma unroll
        for (int it = 0; it < 8; it++) {
          int chunk = it * 32 + lane;
          int d = chunk >> 2, sub = chunk & 3;
          async_copy16(vtn + chunk * 16,
                       Vglob + ((size_t)d * TDIM + nk) * 2 + sub * 16);
        }
      }
      if (wid < 2) asm volatile("s_wait_asynccnt 0x8" ::: "memory");
    } else {
      if (wid < 2) asm volatile("s_wait_asynccnt 0x0" ::: "memory");
    }
    __syncthreads();  // current buffer ready for all waves

    // S^T tiles: keys (rows) x queries (cols). 4 WMMAs from LDS A-frags.
    v16bf ka00 = load_a(Kt + (0 + lq) * 64, 0, lane);
    v16bf ka01 = load_a(Kt + (0 + lq) * 64, 32, lane);
    v16bf ka10 = load_a(Kt + (16 + lq) * 64, 0, lane);
    v16bf ka11 = load_a(Kt + (16 + lq) * 64, 32, lane);

    v8f z = (v8f){0, 0, 0, 0, 0, 0, 0, 0};
    v8f st[2];
    st[0] = wmma_bf16(ka00, qb_0, z);
    st[0] = wmma_bf16(ka01, qb_1, st[0]);
    st[1] = wmma_bf16(ka10, qb_0, z);
    st[1] = wmma_bf16(ka11, qb_1, st[1]);

    // combined causal+attention validity bitmask for this 32-key block
    int dlt = q_abs - key0;
    unsigned cm = (dlt >= 31) ? 0xFFFFFFFFu
                              : ((dlt < 0) ? 0u : ((2u << dlt) - 1u));
    unsigned vb = mrow[jb] & cm;

    // raw max (valid upper bound) -> no pre-max masking needed
    float mx = -3.0e38f;
#pragma unroll
    for (int t = 0; t < 2; t++)
#pragma unroll
      for (int r = 0; r < 8; r++) mx = fmaxf(mx, st[t][r]);
    mx = fmaxf(mx, __shfl_xor(mx, 16));  // merge the two key half-ranges
    float mnew = fmaxf(m_cur, mx);

    float sum = 0.0f;
#pragma unroll
    for (int t = 0; t < 2; t++) {
      float e[8];
#pragma unroll
      for (int r = 0; r < 8; r++) {
        int kloc = t * 16 + r + hi8;
        float ev = __expf(st[t][r] - mnew);
        ev = ((vb >> kloc) & 1u) ? ev : 0.0f;  // mask after exp
        e[r] = ev;
        sum += ev;
      }
      uint4 pk;
      pk.x = pack2bf(e[0], e[1]);
      pk.y = pack2bf(e[2], e[3]);
      pk.z = pack2bf(e[4], e[5]);
      pk.w = pack2bf(e[6], e[7]);
      *(uint4*)(&Plds[wid][lq * 32 + t * 16 + hi8]) = pk;
    }
    sum += __shfl_xor(sum, 16);

    float alpha = __expf(m_cur - mnew);
    m_cur = mnew;
    l_cur = l_cur * alpha + sum;
#pragma unroll
    for (int c = 0; c < 4; c++) acc[c] = acc[c] * alpha;  // per-lane scalar

    asm volatile("" ::: "memory");  // same-wave LDS ops are in-order; pin order
    v16bf pb = load_b(&Plds[wid][lq * 32], 0, lane);

    // acc^T += V^T @ P^T : 4 WMMAs, V^T A-frags from LDS
#pragma unroll
    for (int c = 0; c < 4; c++)
      acc[c] = wmma_bf16(load_a(Vx + (c * 16 + lq) * 32, 0, lane), pb, acc[c]);

    __syncthreads();  // current buffer free before it is re-staged
  }

  // epilogue: lane owns query q0+lq; d = c*16 + hi8 + r -> 16B packed stores
  float inv = 1.0f / fmaxf(l_cur, 1e-20f);
  unsigned short* yrow = yatt + (size_t)(b * TDIM + q0 + lq) * CDIM + h * DDIM;
#pragma unroll
  for (int c = 0; c < 4; c++) {
    uint4 pk;
    pk.x = pack2bf(acc[c][0] * inv, acc[c][1] * inv);
    pk.y = pack2bf(acc[c][2] * inv, acc[c][3] * inv);
    pk.z = pack2bf(acc[c][4] * inv, acc[c][5] * inv);
    pk.w = pack2bf(acc[c][6] * inv, acc[c][7] * inv);
    *(uint4*)(yrow + c * 16 + hi8) = pk;
  }
}

// ---------------- launcher ----------------

extern "C" void kernel_launch(void* const* d_in, const int* in_sizes, int n_in,
                              void* d_out, int out_size, void* d_ws, size_t ws_size,
                              hipStream_t stream) {
  const float* x  = (const float*)d_in[0];
  const int* am   = (const int*)d_in[1];
  const float* Wq = (const float*)d_in[2];
  const float* bq = (const float*)d_in[3];
  const float* Wk = (const float*)d_in[4];
  const float* bk = (const float*)d_in[5];
  const float* Wv = (const float*)d_in[6];
  const float* bv = (const float*)d_in[7];
  const float* Wp = (const float*)d_in[8];
  const float* bp = (const float*)d_in[9];

  char* w = (char*)d_ws;
  unsigned short* xb    = (unsigned short*)(w);                  // 16 MB
  unsigned short* wqkvt = (unsigned short*)(w + 16777216);       // 6 MB
  unsigned short* wpt   = (unsigned short*)(w + 23068672);       // 2 MB
  float*          bqkv  = (float*)(w + 25165824);                // 16 KB slot
  unsigned short* qkv   = (unsigned short*)(w + 25182208);       // 48 MB
  unsigned short* Qh    = (unsigned short*)(w + 75513856);       // 16 MB
  unsigned short* Khp   = (unsigned short*)(w + 92291072);       // 16 MB
  unsigned short* Vtp   = (unsigned short*)(w + 109068288);      // 16 MB
  unsigned short* yatt  = (unsigned short*)(w + 125845504);      // 16 MB
  unsigned int*   mbits = (unsigned int*)(w + 142622720);        // 1 KB

  int nx = MROWS * CDIM;  // 8388608
  cvt_f32_bf16<<<nx / 256, 256, 0, stream>>>(x, xb, nx);
  transpose_cvt<<<(1024 * 1024) / 256, 256, 0, stream>>>(Wq, wqkvt);
  transpose_cvt<<<(1024 * 1024) / 256, 256, 0, stream>>>(Wk, wqkvt + 1024 * 1024);
  transpose_cvt<<<(1024 * 1024) / 256, 256, 0, stream>>>(Wv, wqkvt + 2 * 1024 * 1024);
  transpose_cvt<<<(1024 * 1024) / 256, 256, 0, stream>>>(Wp, wpt);
  concat_bias<<<12, 256, 0, stream>>>(bq, bk, bv, bqkv);
  mask_bits<<<1, 256, 0, stream>>>(am, mbits);

  // fused QKV projection: [8192,1024] x [1024,3072] -> bf16
  gemm_wmma<true><<<dim3(MROWS / 64, 3072 / 256), 256, 0, stream>>>(
      xb, wqkvt, bqkv, (void*)qkv, CDIM, 3072);

  split_heads<<<nx / 256, 256, 0, stream>>>(qkv, Qh, Khp, Vtp);

  // 1024 blocks: each = one head x 128 queries (8 waves x 16)
  flash_attn<<<1024, 256, 0, stream>>>(Qh, Khp, Vtp, mbits, yatt);

  // output projection: [8192,1024] x [1024,1024] + bp -> f32 d_out
  gemm_wmma<false><<<dim3(MROWS / 64, CDIM / 256), 256, 0, stream>>>(
      yatt, wpt, bp, d_out, CDIM, CDIM);
}
